// ResidualVectorQuantizationVanilla_65558380806526
// MI455X (gfx1250) — compile-verified, matrix operand order checked by reading, not run on hardware
//
#include <hip/hip_runtime.h>

// ---------------- problem constants ----------------
#define NROWS 16384      // 8*2048
#define DIM   512
#define KCB   4096
#define NSTAGE 4

// ---------------- tiling ----------------
#define BM 64            // rows per workgroup
#define BN 128           // codebook columns per k-block
#define DC 128           // D chunk staged in LDS
#define LDA (DC + 8)     // padded LDS row stride (ushorts) -> 272B, conflict-free

typedef __attribute__((ext_vector_type(16))) __bf16 v16bf;
typedef __attribute__((ext_vector_type(8)))  float  v8f;
typedef int v4i __attribute__((__vector_size__(4 * sizeof(int))));

// ---- CDNA5 async global->LDS copy (ASYNCcnt-tracked), with sync fallback ----
#if defined(__has_builtin)
#if __has_builtin(__builtin_amdgcn_global_load_async_to_lds_b128)
#define HAVE_ASYNC_LDS 1
#endif
#endif

#ifdef HAVE_ASYNC_LDS
__device__ __forceinline__ void cp16_g2l(const void* g, void* l) {
    __builtin_amdgcn_global_load_async_to_lds_b128(
        (__attribute__((address_space(1))) v4i*)g,
        (__attribute__((address_space(3))) v4i*)l, 0, 0);
}
__device__ __forceinline__ void wait_async() {
#if __has_builtin(__builtin_amdgcn_s_wait_asynccnt)
    __builtin_amdgcn_s_wait_asynccnt(0);
#else
    asm volatile("s_wait_asynccnt 0x0" ::: "memory");
#endif
}
#else
__device__ __forceinline__ void cp16_g2l(const void* g, void* l) {
    *(uint4*)l = *(const uint4*)g;
}
__device__ __forceinline__ void wait_async() {}
#endif

__device__ __forceinline__ unsigned short f2bf(float f) {
    unsigned u = __float_as_uint(f);
    unsigned r = (u + 0x7FFFu + ((u >> 16) & 1u)) >> 16;
    return (unsigned short)r;
}

// ---- prep: codebook f32 -> bf16, plus ||w||^2 per code ----
__global__ __launch_bounds__(256) void rvq_prep_codebook(
    const float* __restrict__ cb, unsigned short* __restrict__ cbBf,
    float* __restrict__ cbSq)
{
    int row = blockIdx.x;          // 0..4095
    int tid = threadIdx.x;         // 0..255, 2 elems each
    float v0 = cb[row * DIM + tid * 2 + 0];
    float v1 = cb[row * DIM + tid * 2 + 1];
    cbBf[row * DIM + tid * 2 + 0] = f2bf(v0);
    cbBf[row * DIM + tid * 2 + 1] = f2bf(v1);
    __shared__ float red[256];
    red[tid] = v0 * v0 + v1 * v1;
    __syncthreads();
    for (int s = 128; s > 0; s >>= 1) {
        if (tid < s) red[tid] += red[tid + s];
        __syncthreads();
    }
    if (tid == 0) cbSq[row] = red[0];
}

// ---- prep: inputs -> residual (f32 + bf16); also zero vq_loss ----
__global__ __launch_bounds__(256) void rvq_prep_residual(
    const float* __restrict__ inp, float* __restrict__ resF,
    unsigned short* __restrict__ resBf, float* __restrict__ vqLoss)
{
    int i = blockIdx.x * 256 + threadIdx.x;        // one float4 each
    float4 v = ((const float4*)inp)[i];
    ((float4*)resF)[i] = v;
    uint2 p;
    p.x = (unsigned)f2bf(v.x) | ((unsigned)f2bf(v.y) << 16);
    p.y = (unsigned)f2bf(v.z) | ((unsigned)f2bf(v.w) << 16);
    ((uint2*)resBf)[i] = p;
    if (blockIdx.x == 0 && threadIdx.x == 0) vqLoss[0] = 0.0f;
}

// ---- one RVQ stage: bf16 WMMA scores + argmin + residual/quant update ----
__global__ __launch_bounds__(256) void rvq_stage(
    const unsigned short* __restrict__ cbBf, const float* __restrict__ cbSq,
    const float* __restrict__ cbF32,
    float* __restrict__ resF, unsigned short* __restrict__ resBf,
    float* __restrict__ quantOut, int* __restrict__ idxOut, int stage)
{
    __shared__ __align__(16) unsigned short lA[BM][LDA];      // 64*136*2  = 17408 B
    // B tile (128*136*2 = 34816 B) aliased with score tile (64*129*4 = 33024 B)
    __shared__ __align__(16) union ShU {
        unsigned short b[BN][LDA];
        float          s[BM][BN + 1];
    } sh;
    __shared__ float lRedV[BM][4];
    __shared__ int   lRedI[BM][4];
    __shared__ float lBestV[BM];
    __shared__ int   lBestI[BM];

    const int tid  = threadIdx.x;
    const int lane = tid & 31;
    const int wid  = tid >> 5;            // 0..7
    const int rowGroup = wid >> 2;        // 0..1 : 32-row slab
    const int colGroup = wid & 3;         // 0..3 : 32-col slab
    const int half = lane >> 4;           // 0/1
    const int m    = lane & 15;
    const int rowBase = blockIdx.x * BM;
    const int rA0 = rowGroup * 32 + m;    // LDS rows for the two A frags
    const int rA1 = rA0 + 16;
    const int cB0 = colGroup * 32 + m;    // LDS rows for the two B frags
    const int cB1 = cB0 + 16;

    if (tid < BM) { lBestV[tid] = 3.4e38f; lBestI[tid] = 0; }

    for (int cbk = 0; cbk < KCB / BN; ++cbk) {
        const int colBase = cbk * BN;

        v8f acc00, acc01, acc10, acc11;
#pragma unroll
        for (int j = 0; j < 8; ++j) {
            acc00[j] = 0.0f; acc01[j] = 0.0f; acc10[j] = 0.0f; acc11[j] = 0.0f;
        }

        // prefetch next column block of the bf16 codebook
        if (cbk + 1 < KCB / BN) {
            const unsigned short* pf =
                cbBf + (size_t)(colBase + BN + (tid & 127)) * DIM + (tid >> 7) * 256;
            __builtin_prefetch(pf, 0, 1);
        }

#pragma unroll 1
        for (int ch = 0; ch < DIM / DC; ++ch) {
            __syncthreads();
            // stage A (32 rows x 128 d) and B (64 codes x 128 d) into LDS
            // via CDNA5 async global->LDS DMA (2 + 4 x b128 per thread)
#pragma unroll
            for (int q = 0; q < 2; ++q) {
                int i = q * 256 + tid;        // 0..511 uint4
                int r = i >> 4;
                int off = (i & 15) * 8;       // ushort offset, 16B aligned
                cp16_g2l(resBf + (size_t)(rowBase + r) * DIM + ch * DC + off,
                         &lA[r][off]);
            }
#pragma unroll
            for (int q = 0; q < 4; ++q) {
                int i = q * 256 + tid;        // 0..1023 uint4
                int r = i >> 4;
                int off = (i & 15) * 8;
                cp16_g2l(cbBf + (size_t)(colBase + r) * DIM + ch * DC + off,
                         &sh.b[r][off]);
            }
            wait_async();
            __syncthreads();

#pragma unroll
            for (int ks = 0; ks < DC / 32; ++ks) {
                const int d0 = ks * 32;
                v16bf a0, a1, b0, b1;
                // 16x32 bf16 frags: two 16B LDS reads each (documented K striping)
                ((uint4*)&a0)[0] = *(const uint4*)&lA[rA0][d0 + half * 8];
                ((uint4*)&a0)[1] = *(const uint4*)&lA[rA0][d0 + 16 + half * 8];
                ((uint4*)&a1)[0] = *(const uint4*)&lA[rA1][d0 + half * 8];
                ((uint4*)&a1)[1] = *(const uint4*)&lA[rA1][d0 + 16 + half * 8];
                ((uint4*)&b0)[0] = *(const uint4*)&sh.b[cB0][d0 + half * 8];
                ((uint4*)&b0)[1] = *(const uint4*)&sh.b[cB0][d0 + 16 + half * 8];
                ((uint4*)&b1)[0] = *(const uint4*)&sh.b[cB1][d0 + half * 8];
                ((uint4*)&b1)[1] = *(const uint4*)&sh.b[cB1][d0 + 16 + half * 8];
                acc00 = __builtin_amdgcn_wmma_f32_16x16x32_bf16(
                    false, a0, false, b0, (short)0, acc00, false, false);
                acc01 = __builtin_amdgcn_wmma_f32_16x16x32_bf16(
                    false, a0, false, b1, (short)0, acc01, false, false);
                acc10 = __builtin_amdgcn_wmma_f32_16x16x32_bf16(
                    false, a1, false, b0, (short)0, acc10, false, false);
                acc11 = __builtin_amdgcn_wmma_f32_16x16x32_bf16(
                    false, a1, false, b1, (short)0, acc11, false, false);
            }
        }

        // all waves done reading B tile before overwriting aliased score tile
        __syncthreads();

        // distances: cb_sq[k] - 2 * <r, w_k>   (||r||^2 constant per row -> dropped)
        {
            const int nl0 = colGroup * 32 + m;
            const int nl1 = nl0 + 16;
            const float cs0 = cbSq[colBase + nl0];
            const float cs1 = cbSq[colBase + nl1];
#pragma unroll
            for (int j = 0; j < 8; ++j) {
                int r0 = rowGroup * 32 + half * 8 + j;  // C layout: vgpr j -> M=j / j+8
                int r1 = r0 + 16;
                sh.s[r0][nl0] = cs0 - 2.0f * acc00[j];
                sh.s[r0][nl1] = cs1 - 2.0f * acc01[j];
                sh.s[r1][nl0] = cs0 - 2.0f * acc10[j];
                sh.s[r1][nl1] = cs1 - 2.0f * acc11[j];
            }
        }
        __syncthreads();

        // per-row argmin over this 128-col block (first-min tie rule kept)
        {
            int r  = tid >> 2;
            int c0 = (tid & 3) * 32;
            float bv = sh.s[r][c0];
            int   bi = colBase + c0;
#pragma unroll
            for (int c = 1; c < 32; ++c) {
                float v = sh.s[r][c0 + c];
                if (v < bv) { bv = v; bi = colBase + c0 + c; }
            }
            lRedV[r][tid & 3] = bv;
            lRedI[r][tid & 3] = bi;
        }
        __syncthreads();
        if (tid < BM) {
            float bv = lRedV[tid][0];
            int   bi = lRedI[tid][0];
#pragma unroll
            for (int t = 1; t < 4; ++t) {
                float v = lRedV[tid][t];
                int   i = lRedI[tid][t];
                if (v < bv || (v == bv && i < bi)) { bv = v; bi = i; }
            }
            if (bv < lBestV[tid]) { lBestV[tid] = bv; lBestI[tid] = bi; }
        }
        __syncthreads();
    }

    // update: quantized += w[idx]; residual -= w[idx]; refresh bf16 residual
#pragma unroll 1
    for (int ii = 0; ii < 32; ++ii) {
        int e   = ii * 256 + tid;        // 0..8191 float4 (64 rows x 128 f4)
        int r   = e >> 7;
        int off = e & 127;
        int g   = rowBase + r;
        int idx = lBestI[r];
        float4 w  = ((const float4*)cbF32)[(size_t)idx * 128 + off];
        float4 rv = ((float4*)resF)[(size_t)g * 128 + off];
        float4 nr = make_float4(rv.x - w.x, rv.y - w.y, rv.z - w.z, rv.w - w.w);
        ((float4*)resF)[(size_t)g * 128 + off] = nr;
        uint2 p;
        p.x = (unsigned)f2bf(nr.x) | ((unsigned)f2bf(nr.y) << 16);
        p.y = (unsigned)f2bf(nr.z) | ((unsigned)f2bf(nr.w) << 16);
        ((uint2*)resBf)[(size_t)g * 128 + off] = p;
        float4* o = (float4*)quantOut + (size_t)g * 128 + off;
        if (stage == 0) {
            *o = w;
        } else {
            float4 c = *o;
            *o = make_float4(c.x + w.x, c.y + w.y, c.z + w.z, c.w + w.w);
        }
    }
    if (tid < BM) idxOut[(size_t)(rowBase + tid) * NSTAGE + stage] = lBestI[tid];
}

extern "C" void kernel_launch(void* const* d_in, const int* in_sizes, int n_in,
                              void* d_out, int out_size, void* d_ws, size_t ws_size,
                              hipStream_t stream) {
    (void)in_sizes; (void)n_in; (void)out_size; (void)ws_size;
    const float* inp   = (const float*)d_in[0];   // [8,2048,512] f32
    const float* cbF32 = (const float*)d_in[1];   // [4096,512]  f32

    float* out      = (float*)d_out;
    float* quantOut = out;                                         // N*D f32
    int*   idxOut   = (int*)(out + (size_t)NROWS * DIM);           // N*4 int32
    float* vqLoss   = out + (size_t)NROWS * DIM + (size_t)NROWS * NSTAGE;

    char* w = (char*)d_ws;
    float*          resF  = (float*)w;          w += (size_t)NROWS * DIM * 4;
    unsigned short* resBf = (unsigned short*)w; w += (size_t)NROWS * DIM * 2;
    unsigned short* cbBf  = (unsigned short*)w; w += (size_t)KCB * DIM * 2;
    float*          cbSq  = (float*)w;

    rvq_prep_codebook<<<KCB, 256, 0, stream>>>(cbF32, cbBf, cbSq);
    rvq_prep_residual<<<(NROWS * DIM / 4) / 256, 256, 0, stream>>>(inp, resF, resBf, vqLoss);

    for (int s = 0; s < NSTAGE; ++s) {
        rvq_stage<<<NROWS / BM, 256, 0, stream>>>(
            cbBf, cbSq, cbF32, resF, resBf, quantOut, idxOut, s);
    }
}
